// DotProductAttention_22436909154804
// MI455X (gfx1250) — matrix operand order repaired and checked by reading, not compiled, not run on hardware
//
#include <hip/hip_runtime.h>
#include <hip/hip_bf16.h>

// ---------------------------------------------------------------------------
// Fused attention for MI455X (gfx1250, wave32, WMMA):
//   scores = Q K^T / sqrt(D); masked; softmax; ctx = P V; outputs ctx AND P.
// B=64, S=2048, D=64, fp32 in/out. 1 GiB weight write dominates -> memory
// bound; use bf16 WMMA (16x16x32) for both GEMMs, f32 accumulation.
// ---------------------------------------------------------------------------

typedef __attribute__((ext_vector_type(16))) __bf16 v16bf;
typedef __attribute__((ext_vector_type(8)))  float  v8f;

constexpr int BB = 64;
constexpr int SS = 2048;
constexpr int DD = 64;
constexpr int QT = 16;                 // query rows per workgroup
constexpr int NWAVE = 4;               // waves per workgroup
constexpr int KSEC = SS / NWAVE;       // 512 keys per wave

__global__ __launch_bounds__(NWAVE * 32)
void attn_fused_kernel(const float* __restrict__ Q,
                       const float* __restrict__ K,
                       const float* __restrict__ V,
                       const int*   __restrict__ M,
                       float* __restrict__ ctx,
                       float* __restrict__ W)
{
    const int tid  = threadIdx.x;
    const int wave = tid >> 5;
    const int lane = tid & 31;
    const int lrow = lane & 15;        // 0..15  (A row / B,C column)
    const int lhi  = lane >> 4;        // 0..1   (K-half / M-half select)

    const int b  = blockIdx.x / (SS / QT);
    const int q0 = (blockIdx.x % (SS / QT)) * QT;

    const size_t base = (size_t)b * SS * DD;
    const float* Qb = Q + base + (size_t)q0 * DD;
    const float* Kb = K + base;
    const float* Vb = V + base;

    // ---- LDS (≈21 KB total) ----
    __shared__ __bf16 stage[NWAVE][QT][32];     // per-wave P staging (A frag)
    __shared__ float  smax[NWAVE][QT];
    __shared__ float  ssum[NWAVE][QT];
    __shared__ float  red[NWAVE][QT][DD];       // partial-context reduction

    // ---- Q tile as two A-fragments (D=64 -> 2 chunks of K=32), bf16 ----
    v16bf qa[2];
    {
        const float* qr = Qb + (size_t)lrow * DD;
        #pragma unroll
        for (int c = 0; c < 2; ++c) {
            #pragma unroll
            for (int e = 0; e < 16; ++e) {
                int kk = 32 * c + 16 * lhi + e;  // contiguous k order (matches B)
                qa[c][e] = (__bf16)qr[kk];
            }
        }
    }

    // =================== Pass A: online softmax statistics ===================
    float m8[8], l8[8];
    #pragma unroll
    for (int r = 0; r < 8; ++r) { m8[r] = -3.0e38f; l8[r] = 0.0f; }

    const int k0 = wave * KSEC;
    for (int t = 0; t < KSEC / 16; ++t) {              // 32 key tiles / wave
        const int kb = k0 + t * 16;
        const float* kr = Kb + (size_t)(kb + lrow) * DD;
        if (t + 1 < KSEC / 16)                          // global_prefetch_b8
            __builtin_prefetch(Kb + (size_t)(kb + 16 + lrow) * DD, 0, 0);

        v16bf b0, b1;                                   // B frags: K^T chunks
        #pragma unroll
        for (int e = 0; e < 16; ++e) {
            b0[e] = (__bf16)kr[      16 * lhi + e];
            b1[e] = (__bf16)kr[32 + 16 * lhi + e];
        }
        v8f acc = {};
        acc = __builtin_amdgcn_wmma_f32_16x16x32_bf16(false, qa[0], false, b0,
                                                      (short)0, acc, false, false);
        acc = __builtin_amdgcn_wmma_f32_16x16x32_bf16(false, qa[1], false, b1,
                                                      (short)0, acc, false, false);
        const int kcol = kb + lrow;
        #pragma unroll
        for (int r = 0; r < 8; ++r) {
            const int qq = r + 8 * lhi;
            float s = acc[r] * 0.125f;                  // 1/sqrt(64)
            const int mk = M[(size_t)(q0 + qq) * SS + kcol];
            s = (mk == 0) ? -1.0e9f : s;
            const float mn = fmaxf(m8[r], s);
            l8[r] = l8[r] * __expf(m8[r] - mn) + __expf(s - mn);
            m8[r] = mn;
        }
    }
    // half-wave (16-lane) merge of (max, sum) pairs
    #pragma unroll
    for (int off = 1; off < 16; off <<= 1) {
        #pragma unroll
        for (int r = 0; r < 8; ++r) {
            const float om = __shfl_xor(m8[r], off, 32);
            const float ol = __shfl_xor(l8[r], off, 32);
            const float mn = fmaxf(m8[r], om);
            l8[r] = l8[r] * __expf(m8[r] - mn) + ol * __expf(om - mn);
            m8[r] = mn;
        }
    }
    if (lrow == 0) {
        #pragma unroll
        for (int r = 0; r < 8; ++r) {
            smax[wave][8 * lhi + r] = m8[r];
            ssum[wave][8 * lhi + r] = l8[r];
        }
    }
    __syncthreads();

    float fm[8], finv[8];                               // final row stats
    #pragma unroll
    for (int r = 0; r < 8; ++r) {
        const int qq = r + 8 * lhi;
        float mn = -3.0e38f, ls = 0.0f;
        #pragma unroll
        for (int w2 = 0; w2 < NWAVE; ++w2) {
            const float om = smax[w2][qq], ol = ssum[w2][qq];
            const float m2 = fmaxf(mn, om);
            ls = ls * __expf(mn - m2) + ol * __expf(om - m2);
            mn = m2;
        }
        fm[r]   = mn;
        finv[r] = 1.0f / ls;
    }

    // ========= Pass B: recompute scores -> P (write) -> ctx += P*V ==========
    v8f cacc[4] = {};                                   // 4 N-tiles of D=64
    float* Wrow = W + (size_t)b * SS * SS + (size_t)q0 * SS;

    for (int c = 0; c < KSEC / 32; ++c) {               // 16 chunks of 32 keys
        const int kb = k0 + c * 32;
        #pragma unroll
        for (int half = 0; half < 2; ++half) {
            const int kt = kb + 16 * half;
            const float* kr = Kb + (size_t)(kt + lrow) * DD;
            v16bf b0, b1;
            #pragma unroll
            for (int e = 0; e < 16; ++e) {
                b0[e] = (__bf16)kr[      16 * lhi + e];
                b1[e] = (__bf16)kr[32 + 16 * lhi + e];
            }
            v8f acc = {};
            acc = __builtin_amdgcn_wmma_f32_16x16x32_bf16(false, qa[0], false, b0,
                                                          (short)0, acc, false, false);
            acc = __builtin_amdgcn_wmma_f32_16x16x32_bf16(false, qa[1], false, b1,
                                                          (short)0, acc, false, false);
            const int kcol = kt + lrow;
            #pragma unroll
            for (int r = 0; r < 8; ++r) {
                const int qq = r + 8 * lhi;
                float s = acc[r] * 0.125f;
                const int mk = M[(size_t)(q0 + qq) * SS + kcol];
                s = (mk == 0) ? -1.0e9f : s;
                const float p = __expf(s - fm[r]) * finv[r];
                Wrow[(size_t)qq * SS + kcol] = p;       // fp32 weights out
                stage[wave][qq][16 * half + lrow] = (__bf16)p;
            }
        }
        // P A-fragment from per-wave staging (same-wave DS ops are in order)
        v16bf wa;
        #pragma unroll
        for (int e = 0; e < 16; ++e) {
            const int kk = 16 * lhi + e;
            wa[e] = stage[wave][lrow][kk];
        }
        // V as B-fragments; accumulate the 4 column tiles of D
        #pragma unroll
        for (int n = 0; n < 4; ++n) {
            v16bf vb;
            #pragma unroll
            for (int e = 0; e < 16; ++e)
                vb[e] = (__bf16)Vb[(size_t)(kb + 16 * lhi + e) * DD + 16 * n + lrow];
            cacc[n] = __builtin_amdgcn_wmma_f32_16x16x32_bf16(false, wa, false, vb,
                                                              (short)0, cacc[n],
                                                              false, false);
        }
    }

    // ---- cross-wave reduction of partial contexts ----
    #pragma unroll
    for (int n = 0; n < 4; ++n)
        #pragma unroll
        for (int r = 0; r < 8; ++r)
            red[wave][r + 8 * lhi][16 * n + lrow] = cacc[n][r];
    __syncthreads();

    float* Cb = ctx + base + (size_t)q0 * DD;
    for (int i = tid; i < QT * DD; i += NWAVE * 32) {
        const int qq = i >> 6, dd = i & 63;
        Cb[(size_t)qq * DD + dd] =
            red[0][qq][dd] + red[1][qq][dd] + red[2][qq][dd] + red[3][qq][dd];
    }
}

extern "C" void kernel_launch(void* const* d_in, const int* in_sizes, int n_in,
                              void* d_out, int out_size, void* d_ws, size_t ws_size,
                              hipStream_t stream) {
    const float* Q = (const float*)d_in[0];
    const float* K = (const float*)d_in[1];
    const float* V = (const float*)d_in[2];
    const int*   M = (const int*)d_in[3];

    float* ctx = (float*)d_out;                                   // [B,S,D]
    float* W   = (float*)d_out + (size_t)BB * SS * DD;            // [B,S,S]

    const int grid = BB * (SS / QT);                              // 8192 WGs
    attn_fused_kernel<<<grid, NWAVE * 32, 0, stream>>>(Q, K, V, M, ctx, W);
}
